// EmotionGATv2_39505109188781
// MI455X (gfx1250) — matrix operand order repaired and checked by reading, not compiled, not run on hardware
//
#include <hip/hip_runtime.h>
#include <hip/hip_bf16.h>
#include <stdint.h>

// ---------------------------------------------------------------------------
// Problem constants (match reference)
// ---------------------------------------------------------------------------
#define NN      65536
#define EE      524288
#define F_IN    64
#define HIDC    64
#define HEADS4  4
#define HC256   256
#define NGRAPH  256
#define NCLS    8

typedef __attribute__((ext_vector_type(16))) __bf16       v16bf;
typedef __attribute__((ext_vector_type(8)))  float        v8f;
typedef __attribute__((ext_vector_type(4)))  unsigned int u32x4;
typedef __attribute__((ext_vector_type(8)))  int          i32x8;
typedef __attribute__((ext_vector_type(4)))  int          i32x4;

// ---------------------------------------------------------------------------
// Helpers
// ---------------------------------------------------------------------------
__device__ __forceinline__ unsigned short f32_to_bf16(float f) {
    unsigned int u = __float_as_uint(f);
    unsigned int r = u + 0x7FFFu + ((u >> 16) & 1u);   // round-to-nearest-even
    return (unsigned short)(r >> 16);
}

__device__ __forceinline__ void atomicMaxFloat(float* addr, float val) {
    if (__float_as_int(val) >= 0)
        atomicMax((int*)addr, __float_as_int(val));
    else
        atomicMin((unsigned int*)addr, (unsigned int)__float_as_int(val));
}

__device__ __forceinline__ float mishf(float v) {
    return v * tanhf(log1pf(expf(v)));
}

// ---------------------------------------------------------------------------
// f32 -> bf16 conversion / fills
// ---------------------------------------------------------------------------
__global__ void cvt_bf16_kernel(const float* __restrict__ x,
                                unsigned short* __restrict__ y, size_t n) {
    size_t i = (size_t)blockIdx.x * blockDim.x + threadIdx.x;
    if (i < n) y[i] = f32_to_bf16(x[i]);
}

__global__ void fill_f32_kernel(float* __restrict__ p, float v, size_t n) {
    size_t i = (size_t)blockIdx.x * blockDim.x + threadIdx.x;
    if (i < n) p[i] = v;
}

__global__ void bias_bcast_kernel(float* __restrict__ out,
                                  const float* __restrict__ bias,
                                  unsigned int mask, size_t n) {
    size_t i = (size_t)blockIdx.x * blockDim.x + threadIdx.x;
    if (i < n) out[i] = bias[i & mask];
}

// ---------------------------------------------------------------------------
// WMMA GEMM with TDM-staged weights:
//   C[M x Nout] = A[M x K](bf16) @ W[K x Nout](bf16) + bias
//
// Block = 256 threads = 8 waves = 4 M-tiles x 2 N-halves.  The whole W
// (K x Nout bf16, <=128KB) is DMA'd into LDS once per block by wave 0 via
// TENSOR_LOAD_TO_LDS (2D D#: data_size=2B, tensor_dim0=Nout, tensor_dim1=K,
// tile=full, stride0=Nout); completion via s_wait_tensorcnt 0 + barrier.
// Each wave owns one 16-row M-tile and one N-half (NT = Nout/32 tiles),
// loads its A fragment once per k-step and sweeps its NT B-tiles out of LDS
// (16-bit reads at stride Nout), accumulating NT v8f accumulators (64 VGPRs max -> no
// scratch spills) with v_wmma_f32_16x16x32_bf16.
//
// Fragment layouts per CDNA5 ISA 7.12.2:
//   A (16x32 bf16): lane m=lane&15; element e -> K = (e&7) + 8*((e>>3)*2 + hi)
//   B (32x16 bf16): lane n=lane&15 holds column N; element e -> same K as A
//   C/D (16x16 f32): VGPR r holds row M = r + 8*hi, column N = lane&15
// ---------------------------------------------------------------------------
template <int NT>
__global__ void gemm_bf16_wmma_kernel(const unsigned short* __restrict__ A,
                                      const unsigned short* __restrict__ W,
                                      const float* __restrict__ bias,
                                      float* __restrict__ C,
                                      int M, int K) {
    constexpr int Nout = NT * 32;
    extern __shared__ unsigned short lds_w[];

    const int warp = threadIdx.x >> 5;
    const int lane = threadIdx.x & 31;

    // ---- stage W[K x Nout] into LDS ----------------------------------------
#if __has_builtin(__builtin_amdgcn_tensor_load_to_lds)
    if (warp == 0) {
        const unsigned long long gaddr = (unsigned long long)(uintptr_t)W;
        const unsigned int lds_off = (unsigned int)(uintptr_t)lds_w; // LDS byte offset
        u32x4 g0;
        g0[0] = 1u;                                      // count=1, user descriptor
        g0[1] = lds_off;                                 // lds_addr
        g0[2] = (unsigned int)gaddr;                     // global_addr[31:0]
        g0[3] = (unsigned int)((gaddr >> 32) & 0x1FFFFFFu) | (2u << 30); // [56:32]|type=2
        i32x8 g1;
        g1[0] = (int)(1u << 16);                         // data_size=1 (2 bytes)
        g1[1] = (int)(((unsigned)Nout & 0xFFFFu) << 16); // tensor_dim0[15:0]
        g1[2] = (int)((((unsigned)Nout >> 16) & 0xFFFFu) |
                      (((unsigned)K & 0xFFFFu) << 16));  // dim0[31:16] | dim1[15:0]
        g1[3] = (int)((((unsigned)K >> 16) & 0xFFFFu) |
                      (((unsigned)Nout & 0xFFFFu) << 16)); // dim1[31:16] | tile_dim0
        g1[4] = (int)((unsigned)K & 0xFFFFu);            // tile_dim1=K, tile_dim2=0
        g1[5] = (int)(unsigned)Nout;                     // tensor_dim0_stride[31:0]
        g1[6] = 0;                                       // stride0[47:32] | stride1[15:0]
        g1[7] = 0;
        const i32x4 z4 = {0, 0, 0, 0};                   // groups 2/3 unused (2D)
        const i32x8 z8 = {0, 0, 0, 0, 0, 0, 0, 0};
        __builtin_amdgcn_tensor_load_to_lds(g0, g1, z4, z4, z8, 0);
        __builtin_amdgcn_s_wait_tensorcnt(0);
    }
#else
    for (int i = threadIdx.x; i < (K * Nout) / 8; i += (int)blockDim.x)
        ((u32x4*)lds_w)[i] = ((const u32x4*)W)[i];
#endif
    __syncthreads();

    // ---- per-wave: one 16-row M tile, one N half ---------------------------
    const int m0 = (blockIdx.x * 4 + (warp >> 1)) << 4;   // 4 M-tiles per block
    if (m0 >= M) return;
    const int n_base = (warp & 1) * (NT * 16);            // N half
    const int hi   = lane >> 4;
    const int col  = lane & 15;
    const int mrow = m0 + col;

    union Frag { v16bf v; u32x4 q[2]; };

    v8f acc[NT];
#pragma unroll
    for (int t = 0; t < NT; ++t) acc[t] = (v8f){};

    for (int k0 = 0; k0 < K; k0 += 32) {
        // A fragment: two 16B contiguous chunks per lane (from global, L2-hot)
        Frag a;
        const unsigned short* ap = A + (size_t)mrow * K + k0 + 8 * hi;
        a.q[0] = *(const u32x4*)(ap);
        a.q[1] = *(const u32x4*)(ap + 16);

        // B fragments from LDS: lane holds W column (n_base+t*16+col), K in A's order
        const unsigned short* wcol = lds_w + (size_t)(k0 + 8 * hi) * Nout + n_base + col;
#pragma unroll
        for (int t = 0; t < NT; ++t) {
            Frag b;
#pragma unroll
            for (int j = 0; j < 8; ++j) {
                const int kk = ((2 * j) & 7) + (((2 * j) >> 3) << 4);
                b.q[j >> 2][j & 3] = (unsigned int)wcol[(size_t)kk * Nout + t * 16] |
                                     ((unsigned int)wcol[(size_t)(kk + 1) * Nout + t * 16] << 16);
            }
            acc[t] = __builtin_amdgcn_wmma_f32_16x16x32_bf16(
                false, a.v, false, b.v, (short)0, acc[t], false, false);
        }
    }

#pragma unroll
    for (int t = 0; t < NT; ++t) {
        const int n0 = n_base + t * 16;
        const float bcol = bias[n0 + col];
#pragma unroll
        for (int r = 0; r < 8; ++r)
            C[(size_t)(m0 + r + 8 * hi) * Nout + n0 + col] = acc[t][r] + bcol;
    }
}

// ---------------------------------------------------------------------------
// Edge kernels (GATv2 attention).  One wave32 per edge.  Self-loops are the
// last NN edges (src = dst = e - EE).
// ---------------------------------------------------------------------------
template <int H, int C>
__global__ void edge_alpha_kernel(const float* __restrict__ xl,
                                  const float* __restrict__ xr,
                                  const int* __restrict__ ei,
                                  const float* __restrict__ att,
                                  float* __restrict__ alpha,
                                  float* __restrict__ amax,
                                  int E, int n) {
    constexpr int HCv   = H * C;
    constexpr int PER   = HCv / 32;
    constexpr int GROUP = 32 / H;
    const int e    = blockIdx.x * (blockDim.x >> 5) + (threadIdx.x >> 5);
    const int lane = threadIdx.x & 31;
    const int Etot = E + n;
    if (e >= Etot) return;
    const int s = (e < E) ? ei[e]     : (e - E);
    const int d = (e < E) ? ei[E + e] : (e - E);

    const float* pl = xl + (size_t)s * HCv + lane * PER;
    const float* pr = xr + (size_t)d * HCv + lane * PER;
    float part = 0.f;
#pragma unroll
    for (int i = 0; i < PER; ++i) {
        float v = pl[i] + pr[i];
        v = (v > 0.f) ? v : 0.2f * v;          // LeakyReLU(0.2)
        part += v * att[lane * PER + i];
    }
#pragma unroll
    for (int off = GROUP >> 1; off > 0; off >>= 1)
        part += __shfl_xor(part, off, 32);

    if ((lane & (GROUP - 1)) == 0) {
        const int h = lane / GROUP;
        alpha[(size_t)e * H + h] = part;
        atomicMaxFloat(&amax[(size_t)d * H + h], part);
    }
}

__global__ void edge_expsum_kernel(const int* __restrict__ ei,
                                   float* __restrict__ alpha,
                                   const float* __restrict__ amax,
                                   float* __restrict__ denom,
                                   int E, int n, int H) {
    const int idx  = blockIdx.x * blockDim.x + threadIdx.x;
    const int Etot = E + n;
    if (idx >= Etot * H) return;
    const int e = idx / H;
    const int h = idx - e * H;
    const int d = (e < E) ? ei[E + e] : (e - E);
    const float w = expf(alpha[idx] - amax[(size_t)d * H + h]);
    alpha[idx] = w;
    atomicAdd(&denom[(size_t)d * H + h], w);
}

template <int H, int C>
__global__ void edge_aggregate_kernel(const float* __restrict__ xl,
                                      const int* __restrict__ ei,
                                      const float* __restrict__ alpha,
                                      const float* __restrict__ denom,
                                      float* __restrict__ out,
                                      int E, int n) {
    constexpr int HCv   = H * C;
    constexpr int PER   = HCv / 32;
    constexpr int GROUP = 32 / H;
    const int e    = blockIdx.x * (blockDim.x >> 5) + (threadIdx.x >> 5);
    const int lane = threadIdx.x & 31;
    const int Etot = E + n;
    if (e >= Etot) return;
    const int s = (e < E) ? ei[e]     : (e - E);
    const int d = (e < E) ? ei[E + e] : (e - E);
    const int h = lane / GROUP;
    const float coef = alpha[(size_t)e * H + h] /
                       (denom[(size_t)d * H + h] + 1e-16f);
    const float* pl = xl + (size_t)s * HCv + lane * PER;
    float*       po = out + (size_t)d * HCv + lane * PER;
#pragma unroll
    for (int i = 0; i < PER; ++i) atomicAdd(po + i, pl[i] * coef);
}

// ---------------------------------------------------------------------------
// BatchNorm: per-column stats, then apply (+ optional mish), emit f32 + bf16
// ---------------------------------------------------------------------------
__global__ void bn_stats_kernel(const float* __restrict__ x,
                                float* __restrict__ mean,
                                float* __restrict__ var,
                                int n, int hc) {
    const int col = blockIdx.x;
    __shared__ float ssum[256];
    __shared__ float ssq[256];
    float s = 0.f, q = 0.f;
    for (int r = threadIdx.x; r < n; r += blockDim.x) {
        const float v = x[(size_t)r * hc + col];
        s += v; q += v * v;
    }
    ssum[threadIdx.x] = s; ssq[threadIdx.x] = q;
    __syncthreads();
    for (int off = 128; off > 0; off >>= 1) {
        if ((int)threadIdx.x < off) {
            ssum[threadIdx.x] += ssum[threadIdx.x + off];
            ssq[threadIdx.x]  += ssq[threadIdx.x + off];
        }
        __syncthreads();
    }
    if (threadIdx.x == 0) {
        const float mu = ssum[0] / n;
        mean[col] = mu;
        var[col]  = ssq[0] / n - mu * mu;
    }
}

__global__ void bn_apply_kernel(const float* __restrict__ x,
                                const float* __restrict__ mean,
                                const float* __restrict__ var,
                                const float* __restrict__ g,
                                const float* __restrict__ beta,
                                float* __restrict__ yf,
                                unsigned short* __restrict__ yb,
                                size_t total, unsigned int mask, int do_mish) {
    const size_t i = (size_t)blockIdx.x * blockDim.x + threadIdx.x;
    if (i >= total) return;
    const unsigned int col = (unsigned int)i & mask;
    float v = (x[i] - mean[col]) * rsqrtf(var[col] + 1e-5f) * g[col] + beta[col];
    if (do_mish) v = mishf(v);
    yf[i] = v;
    yb[i] = f32_to_bf16(v);
}

// ---------------------------------------------------------------------------
// Global attention pooling + classifier
// ---------------------------------------------------------------------------
__global__ void gate_kernel(const float* __restrict__ h,
                            const float* __restrict__ gW1,
                            const float* __restrict__ gb1,
                            const float* __restrict__ gW2,
                            const float* __restrict__ gb2,
                            const int* __restrict__ batch,
                            float* __restrict__ gate,
                            float* __restrict__ gmax, int n) {
    const int nd = blockIdx.x * blockDim.x + threadIdx.x;
    if (nd >= n) return;
    const float* row = h + (size_t)nd * HIDC;
    float acc = gb2[0];
#pragma unroll 4
    for (int j = 0; j < 32; ++j) {
        float t = gb1[j];
        for (int c = 0; c < HIDC; ++c) t += row[c] * gW1[c * 32 + j];
        t = (t > 0.f) ? t : 0.f;
        acc += t * gW2[j];
    }
    gate[nd] = acc;
    atomicMaxFloat(&gmax[batch[nd]], acc);
}

__global__ void pool_w_kernel(const float* __restrict__ gate,
                              const int* __restrict__ batch,
                              const float* __restrict__ gmax,
                              float* __restrict__ w,
                              float* __restrict__ wsum, int n) {
    const int nd = blockIdx.x * blockDim.x + threadIdx.x;
    if (nd >= n) return;
    const int b = batch[nd];
    const float v = expf(gate[nd] - gmax[b]);
    w[nd] = v;
    atomicAdd(&wsum[b], v);
}

__global__ void pool_acc_kernel(const float* __restrict__ h,
                                const int* __restrict__ batch,
                                const float* __restrict__ w,
                                const float* __restrict__ wsum,
                                float* __restrict__ pooled, int n) {
    const int nd = blockIdx.x * blockDim.x + threadIdx.x;
    if (nd >= n) return;
    const int b = batch[nd];
    const float coef = w[nd] / (wsum[b] + 1e-16f);
    const float* row = h + (size_t)nd * HIDC;
    for (int c = 0; c < HIDC; ++c)
        atomicAdd(&pooled[(size_t)b * HIDC + c], row[c] * coef);
}

__global__ void final_mlp_kernel(const float* __restrict__ pooled,
                                 const float* __restrict__ cW1, const float* __restrict__ cb1,
                                 const float* __restrict__ cW2, const float* __restrict__ cb2,
                                 const float* __restrict__ cW3, const float* __restrict__ cb3,
                                 float* __restrict__ out) {
    const int g = blockIdx.x * blockDim.x + threadIdx.x;
    if (g >= NGRAPH) return;
    float z1[128], z2[64];
    const float* p = pooled + (size_t)g * HIDC;
    for (int j = 0; j < 128; ++j) {
        float t = cb1[j];
        for (int c = 0; c < 64; ++c) t += p[c] * cW1[c * 128 + j];
        z1[j] = mishf(t);
    }
    for (int j = 0; j < 64; ++j) {
        float t = cb2[j];
        for (int c = 0; c < 128; ++c) t += z1[c] * cW2[c * 64 + j];
        z2[j] = mishf(t);
    }
    for (int j = 0; j < NCLS; ++j) {
        float t = cb3[j];
        for (int c = 0; c < 64; ++c) t += z2[c] * cW3[c * NCLS + j];
        out[(size_t)g * NCLS + j] = t;
    }
}

// ---------------------------------------------------------------------------
// Host orchestration
// ---------------------------------------------------------------------------
extern "C" void kernel_launch(void* const* d_in, const int* in_sizes, int n_in,
                              void* d_out, int out_size, void* d_ws, size_t ws_size,
                              hipStream_t stream) {
    const float* x     = (const float*)d_in[0];
    const int*   ei    = (const int*)  d_in[1];
    const int*   batch = (const int*)  d_in[2];
    const float* W1l = (const float*)d_in[3],  *b1l = (const float*)d_in[4];
    const float* W1r = (const float*)d_in[5],  *b1r = (const float*)d_in[6];
    const float* att1  = (const float*)d_in[7],  *bias1 = (const float*)d_in[8];
    const float* g1    = (const float*)d_in[9],  *beta1 = (const float*)d_in[10];
    const float* W2l = (const float*)d_in[11], *b2l = (const float*)d_in[12];
    const float* W2r = (const float*)d_in[13], *b2r = (const float*)d_in[14];
    const float* att2  = (const float*)d_in[15], *bias2 = (const float*)d_in[16];
    const float* g2    = (const float*)d_in[17], *beta2 = (const float*)d_in[18];
    const float* W3l = (const float*)d_in[19], *b3l = (const float*)d_in[20];
    const float* W3r = (const float*)d_in[21], *b3r = (const float*)d_in[22];
    const float* att3  = (const float*)d_in[23], *bias3 = (const float*)d_in[24];
    const float* g3    = (const float*)d_in[25], *beta3 = (const float*)d_in[26];
    const float* gW1 = (const float*)d_in[27], *gb1 = (const float*)d_in[28];
    const float* gW2 = (const float*)d_in[29], *gb2 = (const float*)d_in[30];
    const float* cW1 = (const float*)d_in[31], *cb1 = (const float*)d_in[32];
    const float* cW2 = (const float*)d_in[33], *cb2 = (const float*)d_in[34];
    const float* cW3 = (const float*)d_in[35], *cb3 = (const float*)d_in[36];

    const int Etot = EE + NN;

    // ---- workspace carve-up -------------------------------------------------
    char*  ws  = (char*)d_ws;
    size_t off = 0;
    auto take = [&](size_t bytes) -> char* {
        char* p = ws + off;
        off = (off + bytes + 255) & ~(size_t)255;
        return p;
    };
    unsigned short* A16   = (unsigned short*)take((size_t)NN * HC256 * 2); // bf16 input
    float*          XL    = (float*)take((size_t)NN * HC256 * 4);
    float*          XR    = (float*)take((size_t)NN * HC256 * 4);
    float*          OUT   = (float*)take((size_t)NN * HC256 * 4);
    float*          ALPHA = (float*)take((size_t)Etot * HEADS4 * 4);
    float*          AMAX  = (float*)take((size_t)NN * HEADS4 * 4);
    float*          DENOM = (float*)take((size_t)NN * HEADS4 * 4);
    unsigned short* WL16  = (unsigned short*)take((size_t)HC256 * HC256 * 2);
    unsigned short* WR16  = (unsigned short*)take((size_t)HC256 * HC256 * 2);
    float*          MEAN  = (float*)take(HC256 * 4);
    float*          VAR   = (float*)take(HC256 * 4);
    float*          GATE  = (float*)take((size_t)NN * 4);
    float*          WNODE = (float*)take((size_t)NN * 4);
    float*          GMAX  = (float*)take(NGRAPH * 4);
    float*          WSUM  = (float*)take(NGRAPH * 4);
    float*          POOL  = (float*)take((size_t)NGRAPH * HIDC * 4);

    auto gblk = [](size_t n) { return (unsigned int)((n + 255) / 256); };

    auto run_gat_layer = [&](int H, int C, const float* att, const float* bias) {
        const int HCv = H * C;
        fill_f32_kernel<<<gblk((size_t)NN * H), 256, 0, stream>>>(AMAX, -INFINITY, (size_t)NN * H);
        fill_f32_kernel<<<gblk((size_t)NN * H), 256, 0, stream>>>(DENOM, 0.f, (size_t)NN * H);
        const unsigned int eblk = (unsigned int)((Etot + 7) / 8);
        if (H == 4)
            edge_alpha_kernel<4, 64><<<eblk, 256, 0, stream>>>(XL, XR, ei, att, ALPHA, AMAX, EE, NN);
        else
            edge_alpha_kernel<1, 64><<<eblk, 256, 0, stream>>>(XL, XR, ei, att, ALPHA, AMAX, EE, NN);
        edge_expsum_kernel<<<gblk((size_t)Etot * H), 256, 0, stream>>>(ei, ALPHA, AMAX, DENOM, EE, NN, H);
        bias_bcast_kernel<<<gblk((size_t)NN * HCv), 256, 0, stream>>>(OUT, bias, HCv - 1, (size_t)NN * HCv);
        if (H == 4)
            edge_aggregate_kernel<4, 64><<<eblk, 256, 0, stream>>>(XL, ei, ALPHA, DENOM, OUT, EE, NN);
        else
            edge_aggregate_kernel<1, 64><<<eblk, 256, 0, stream>>>(XL, ei, ALPHA, DENOM, OUT, EE, NN);
    };

    auto gemm = [&](const unsigned short* A, const unsigned short* Wp,
                    const float* bias, float* Cout, int M, int K, int Nout) {
        const unsigned int blocks = (unsigned int)(M / 16 / 4); // 4 M-tiles/block
        const size_t shmem = (size_t)K * Nout * 2;              // full W in LDS
        if (Nout == HC256)
            gemm_bf16_wmma_kernel<8><<<blocks, 256, shmem, stream>>>(A, Wp, bias, Cout, M, K);
        else
            gemm_bf16_wmma_kernel<2><<<blocks, 256, shmem, stream>>>(A, Wp, bias, Cout, M, K);
    };

    // ---- layer 1 ------------------------------------------------------------
    cvt_bf16_kernel<<<gblk((size_t)NN * F_IN), 256, 0, stream>>>(x, A16, (size_t)NN * F_IN);
    cvt_bf16_kernel<<<gblk((size_t)F_IN * HC256), 256, 0, stream>>>(W1l, WL16, (size_t)F_IN * HC256);
    cvt_bf16_kernel<<<gblk((size_t)F_IN * HC256), 256, 0, stream>>>(W1r, WR16, (size_t)F_IN * HC256);
    gemm(A16, WL16, b1l, XL, NN, F_IN, HC256);
    gemm(A16, WR16, b1r, XR, NN, F_IN, HC256);
    run_gat_layer(HEADS4, HIDC, att1, bias1);
    bn_stats_kernel<<<HC256, 256, 0, stream>>>(OUT, MEAN, VAR, NN, HC256);
    bn_apply_kernel<<<gblk((size_t)NN * HC256), 256, 0, stream>>>(
        OUT, MEAN, VAR, g1, beta1, XL, A16, (size_t)NN * HC256, HC256 - 1, 1);

    // ---- layer 2 ------------------------------------------------------------
    cvt_bf16_kernel<<<gblk((size_t)HC256 * HC256), 256, 0, stream>>>(W2l, WL16, (size_t)HC256 * HC256);
    cvt_bf16_kernel<<<gblk((size_t)HC256 * HC256), 256, 0, stream>>>(W2r, WR16, (size_t)HC256 * HC256);
    gemm(A16, WL16, b2l, XL, NN, HC256, HC256);
    gemm(A16, WR16, b2r, XR, NN, HC256, HC256);
    run_gat_layer(HEADS4, HIDC, att2, bias2);
    bn_stats_kernel<<<HC256, 256, 0, stream>>>(OUT, MEAN, VAR, NN, HC256);
    bn_apply_kernel<<<gblk((size_t)NN * HC256), 256, 0, stream>>>(
        OUT, MEAN, VAR, g2, beta2, XL, A16, (size_t)NN * HC256, HC256 - 1, 1);

    // ---- layer 3 (1 head, no concat, no mish) -------------------------------
    cvt_bf16_kernel<<<gblk((size_t)HC256 * HIDC), 256, 0, stream>>>(W3l, WL16, (size_t)HC256 * HIDC);
    cvt_bf16_kernel<<<gblk((size_t)HC256 * HIDC), 256, 0, stream>>>(W3r, WR16, (size_t)HC256 * HIDC);
    gemm(A16, WL16, b3l, XL, NN, HC256, HIDC);
    gemm(A16, WR16, b3r, XR, NN, HC256, HIDC);
    run_gat_layer(1, HIDC, att3, bias3);
    bn_stats_kernel<<<HIDC, 256, 0, stream>>>(OUT, MEAN, VAR, NN, HIDC);
    bn_apply_kernel<<<gblk((size_t)NN * HIDC), 256, 0, stream>>>(
        OUT, MEAN, VAR, g3, beta3, XL, A16, (size_t)NN * HIDC, HIDC - 1, 0);

    // ---- global attention pooling ------------------------------------------
    fill_f32_kernel<<<1, 256, 0, stream>>>(GMAX, -INFINITY, NGRAPH);
    fill_f32_kernel<<<1, 256, 0, stream>>>(WSUM, 0.f, NGRAPH);
    fill_f32_kernel<<<gblk((size_t)NGRAPH * HIDC), 256, 0, stream>>>(POOL, 0.f, (size_t)NGRAPH * HIDC);
    gate_kernel<<<gblk(NN), 256, 0, stream>>>(XL, gW1, gb1, gW2, gb2, batch, GATE, GMAX, NN);
    pool_w_kernel<<<gblk(NN), 256, 0, stream>>>(GATE, batch, GMAX, WNODE, WSUM, NN);
    pool_acc_kernel<<<gblk(NN), 256, 0, stream>>>(XL, batch, WNODE, WSUM, POOL, NN);

    // ---- classifier ---------------------------------------------------------
    final_mlp_kernel<<<1, 256, 0, stream>>>(POOL, cW1, cb1, cW2, cb2, cW3, cb3,
                                            (float*)d_out);
}
